// DATMambaLayer_51565377356412
// MI455X (gfx1250) — compile-verified, matrix operand order checked by reading, not run on hardware
//
#include <hip/hip_runtime.h>

typedef __attribute__((ext_vector_type(16))) _Float16 v16h;
typedef __attribute__((ext_vector_type(8)))  float    v8f;
typedef _Float16 half_t;

// ---------------- constants ----------------
#define BB    8
#define L1    196
#define DI1   1024
#define R1    (BB*L1)      // 1568
#define L2    128
#define DM2   784
#define DI2   1568
#define DI2P  1600         // DI2 padded to mult of 64 (as K / as N for dt2)
#define R2    (BB*L2)      // 1024
#define KP2   832          // DM2(784) padded to mult of 64 (as K for in2, as N for out2)
#define NXP1  64           // dtr(32)+2*16
#define XLD2  128          // x_dbl2 ld: 81 padded to mult of 64
#define DTR1  32
#define DTR1P 64
#define DTR2  49
#define DTR2P 64
#define NST   16

// ---------------- block reduction ----------------
__device__ inline float blk_sum(float v, float* sbuf) {
  int lane = threadIdx.x & 31;
  int wid  = threadIdx.x >> 5;
  for (int o = 16; o > 0; o >>= 1) v += __shfl_down(v, o, 32);
  __syncthreads();
  if (lane == 0) sbuf[wid] = v;
  __syncthreads();
  if (threadIdx.x == 0) {
    float r = 0.f;
    int nw = (blockDim.x + 31) >> 5;
    for (int i = 0; i < nw; ++i) r += sbuf[i];
    sbuf[0] = r;
  }
  __syncthreads();
  return sbuf[0];
}

// ---------------- fp32 -> fp16 convert with zero padding ----------------
__global__ void convert_pad(const float* __restrict__ src, int sld, int srows, int scols,
                            half_t* __restrict__ dst, int dld, int drows) {
  int t = blockIdx.x * blockDim.x + threadIdx.x;
  int total = drows * dld;
  if (t >= total) return;
  int r = t / dld, c = t % dld;
  float v = (r < srows && c < scols) ? src[(size_t)r * sld + c] : 0.f;
  dst[t] = (half_t)v;
}

// ---------------- zero-fill padded fp16 columns [c0, ld) ----------------
__global__ void zfill_h(half_t* __restrict__ dst, int ld, int c0, int rows) {
  int w = ld - c0;
  int t = blockIdx.x * blockDim.x + threadIdx.x;
  if (t >= rows * w) return;
  int r = t / w, c = c0 + t % w;
  dst[(size_t)r * ld + c] = (half_t)0.f;
}

// ---------------- WMMA fragment load (ISA 16-bit A/B lane layout) ----------------
__device__ inline v16h load_frag(const half_t* __restrict__ base, int kb, int g) {
  union { v16h v; uint4 q[2]; } u;
  u.q[0] = *(const uint4*)(base + kb + g * 8);
  u.q[1] = *(const uint4*)(base + kb + 16 + g * 8);
  return u.v;
}

// ---------------- epilogue ----------------
__device__ inline void epi_store(v8f a, int m0, int g, int col, int Nld, int Nv,
                                 const float* __restrict__ bias, const float* __restrict__ res,
                                 int act, const float* __restrict__ bnsc,
                                 const float* __restrict__ bnsh,
                                 float* __restrict__ C, half_t* __restrict__ Ch) {
  if (col >= Nv) return;
  float bi = bias ? bias[col] : 0.f;
  float sc = 1.f, sh = 0.f;
  if (act == 3) { sc = bnsc[col]; sh = bnsh[col]; }
#pragma unroll
  for (int v = 0; v < 8; ++v) {
    int row = m0 + v + 8 * g;
    float xv = a[v] + bi;
    if      (act == 1) xv = (xv > 20.f) ? xv : log1pf(expf(xv));
    else if (act == 2) xv = 0.5f * xv * (1.f + erff(xv * 0.70710678f));
    else if (act == 3) { xv = xv * sc + sh; xv = xv < 0.f ? 0.f : xv; }
    if (res) xv += res[(size_t)row * Nld + col];
    size_t o = (size_t)row * Nld + col;
    C[o] = xv;
    if (Ch) Ch[o] = (half_t)xv;
  }
}

// ---------------- WMMA GEMM: C[M,Nld] = A[M,K] * B[Nld,K]^T (+epilogue) ----------------
// Requirements: M % 32 == 0, Nld % 64 == 0, K % 64 == 0. Stores only cols < Nv.
// act: 0 none, 1 softplus, 2 gelu(exact), 3 bn+relu
__global__ void gemm_wmma(const half_t* __restrict__ A, const half_t* __restrict__ Bm,
                          float* __restrict__ C, half_t* __restrict__ Ch,
                          int M, int Nld, int K, int Nv,
                          const float* __restrict__ bias, const float* __restrict__ res,
                          int act, const float* __restrict__ bnsc,
                          const float* __restrict__ bnsh) {
  const int lane = threadIdx.x & 31;
  const int wav  = threadIdx.x >> 5;
  const int rr   = lane & 15;
  const int g    = lane >> 4;
  const int m0   = blockIdx.y * 32 + wav * 16;
  const int n0   = blockIdx.x * 64;

  const half_t* arow  = A  + (size_t)(m0 + rr) * K;
  const half_t* brow0 = Bm + (size_t)(n0 + rr) * K;
  const half_t* brow1 = brow0 + (size_t)16 * K;
  const half_t* brow2 = brow0 + (size_t)32 * K;
  const half_t* brow3 = brow0 + (size_t)48 * K;

  v8f acc0, acc1, acc2, acc3;
#pragma unroll
  for (int i = 0; i < 8; ++i) { acc0[i] = 0.f; acc1[i] = 0.f; acc2[i] = 0.f; acc3[i] = 0.f; }

  for (int kb = 0; kb < K; kb += 64) {
    if (kb + 64 < K) {
      __builtin_prefetch(arow  + kb + 64, 0, 3);  // global_prefetch_b8 (WGP scope)
      __builtin_prefetch(brow0 + kb + 64, 0, 3);
    }
    // Issue ALL 20 fragment loads (both K sub-steps) before any WMMA: loads
    // return in order, so the first 4 WMMAs only wait for the first 10 loads
    // while the second set is still in flight -> real load/WMMA overlap.
    v16h a0  = load_frag(arow,  kb, g);
    v16h b00 = load_frag(brow0, kb, g);
    v16h b01 = load_frag(brow1, kb, g);
    v16h b02 = load_frag(brow2, kb, g);
    v16h b03 = load_frag(brow3, kb, g);
    v16h a1  = load_frag(arow,  kb + 32, g);
    v16h b10 = load_frag(brow0, kb + 32, g);
    v16h b11 = load_frag(brow1, kb + 32, g);
    v16h b12 = load_frag(brow2, kb + 32, g);
    v16h b13 = load_frag(brow3, kb + 32, g);
    acc0 = __builtin_amdgcn_wmma_f32_16x16x32_f16(false, a0, false, b00, (short)0, acc0, false, false);
    acc1 = __builtin_amdgcn_wmma_f32_16x16x32_f16(false, a0, false, b01, (short)0, acc1, false, false);
    acc2 = __builtin_amdgcn_wmma_f32_16x16x32_f16(false, a0, false, b02, (short)0, acc2, false, false);
    acc3 = __builtin_amdgcn_wmma_f32_16x16x32_f16(false, a0, false, b03, (short)0, acc3, false, false);
    acc0 = __builtin_amdgcn_wmma_f32_16x16x32_f16(false, a1, false, b10, (short)0, acc0, false, false);
    acc1 = __builtin_amdgcn_wmma_f32_16x16x32_f16(false, a1, false, b11, (short)0, acc1, false, false);
    acc2 = __builtin_amdgcn_wmma_f32_16x16x32_f16(false, a1, false, b12, (short)0, acc2, false, false);
    acc3 = __builtin_amdgcn_wmma_f32_16x16x32_f16(false, a1, false, b13, (short)0, acc3, false, false);
  }

  epi_store(acc0, m0, g, n0 + rr,      Nld, Nv, bias, res, act, bnsc, bnsh, C, Ch);
  epi_store(acc1, m0, g, n0 + rr + 16, Nld, Nv, bias, res, act, bnsc, bnsh, C, Ch);
  epi_store(acc2, m0, g, n0 + rr + 32, Nld, Nv, bias, res, act, bnsc, bnsh, C, Ch);
  epi_store(acc3, m0, g, n0 + rr + 48, Nld, Nv, bias, res, act, bnsc, bnsh, C, Ch);
}

// ---------------- LN of concat(x1..x4) over 512, fp16 out ----------------
__global__ void ln_concat(const float* __restrict__ x1, const float* __restrict__ x2,
                          const float* __restrict__ x3, const float* __restrict__ x4,
                          const float* __restrict__ g, const float* __restrict__ b,
                          half_t* __restrict__ out) {
  __shared__ float sbuf[32];
  int row = blockIdx.x;
  const float* xs[4] = {x1, x2, x3, x4};
  int c0 = threadIdx.x, c1 = threadIdx.x + 256;
  float v0 = xs[c0 >> 7][(size_t)row * 128 + (c0 & 127)];
  float v1 = xs[c1 >> 7][(size_t)row * 128 + (c1 & 127)];
  float mu = blk_sum(v0 + v1, sbuf) * (1.f / 512.f);
  float d0 = v0 - mu, d1 = v1 - mu;
  float inv = rsqrtf(blk_sum(d0 * d0 + d1 * d1, sbuf) * (1.f / 512.f) + 1e-5f);
  out[(size_t)row * 512 + c0] = (half_t)(d0 * inv * g[c0] + b[c0]);
  out[(size_t)row * 512 + c1] = (half_t)(d1 * inv * g[c1] + b[c1]);
}

// ---------------- build c2 (regroup of m1) + LN over 784, pad to KP2 ----------------
__global__ void ln_c2(const float* __restrict__ m1, const float* __restrict__ g,
                      const float* __restrict__ b, half_t* __restrict__ out) {
  __shared__ float sbuf[32];
  int row = blockIdx.x;  // b*128 + c
  int bb = row >> 7, c = row & 127;
  float v[4]; int j[4];
  float s = 0.f;
  for (int k = 0; k < 4; ++k) {
    j[k] = threadIdx.x + 256 * k;
    if (j[k] < 784) {
      int i = j[k] / 196, l = j[k] % 196;
      v[k] = m1[((size_t)(bb * 196 + l)) * 512 + i * 128 + c];
      s += v[k];
    } else v[k] = 0.f;
  }
  float mu = blk_sum(s, sbuf) * (1.f / 784.f);
  float s2 = 0.f;
  for (int k = 0; k < 4; ++k) if (j[k] < 784) { float d = v[k] - mu; s2 += d * d; }
  float inv = rsqrtf(blk_sum(s2, sbuf) * (1.f / 784.f) + 1e-5f);
  for (int k = 0; k < 4; ++k)
    if (j[k] < KP2)
      out[(size_t)row * KP2 + j[k]] =
          (j[k] < 784) ? (half_t)((v[k] - mu) * inv * g[j[k]] + b[j[k]]) : (half_t)0.f;
}

// ---------------- depthwise causal conv (K=4) + SiLU, fp32(ld=di) + fp16(ld=ldh) ----------------
__global__ void conv_silu(const float* __restrict__ xz, int ld, int di, int L,
                          const float* __restrict__ w, const float* __restrict__ cb,
                          float* __restrict__ xc, half_t* __restrict__ xch, int ldh, int total) {
  int t = blockIdx.x * blockDim.x + threadIdx.x;
  if (t >= total) return;
  int d = t % di;
  int bl = t / di;
  int l = bl % L, b = bl / L;
  float s = cb[d];
#pragma unroll
  for (int k = 0; k < 4; ++k) {
    int ll = l + k - 3;
    if (ll >= 0) s += xz[(size_t)(b * L + ll) * ld + d] * w[d * 4 + k];
  }
  float r = s / (1.f + expf(-s));
  xc[(size_t)bl * di + d] = r;
  xch[(size_t)bl * ldh + d] = (half_t)r;
}

// ---------------- selective scan + D skip + SiLU(z) gate -> fp16 ----------------
__global__ void sel_scan(const float* __restrict__ xc, const float* __restrict__ dt, int dtld,
                         const float* __restrict__ xdbl, int xld, int bOff, int cOff,
                         const float* __restrict__ Alog, const float* __restrict__ Dp,
                         const float* __restrict__ xz, int zld, int zOff,
                         half_t* __restrict__ yh, int yld, int di, int L, int nB) {
  int t = blockIdx.x * blockDim.x + threadIdx.x;
  if (t >= nB * di) return;
  int d = t % di, b = t / di;
  float a[NST], h[NST];
#pragma unroll
  for (int n = 0; n < NST; ++n) { a[n] = -expf(Alog[d * NST + n]); h[n] = 0.f; }
  float Dd = Dp[d];
  for (int l = 0; l < L; ++l) {
    int row = b * L + l;
    float dtv = dt[(size_t)row * dtld + d];
    float u   = xc[(size_t)row * di + d];
    float du  = dtv * u;
    const float* xd = xdbl + (size_t)row * xld;
    float y = 0.f;
#pragma unroll
    for (int n = 0; n < NST; ++n) {
      h[n] = expf(dtv * a[n]) * h[n] + du * xd[bOff + n];
      y += h[n] * xd[cOff + n];
    }
    float zv = xz[(size_t)row * zld + zOff + d];
    float out = (y + u * Dd) * (zv / (1.f + expf(-zv)));
    yh[(size_t)row * yld + d] = (half_t)out;
  }
}

// ---------------- branch: r = x + s(m2 slice), LN(eps 1e-6) -> fp16 ----------------
__global__ void branch_prep(const float* __restrict__ x, const float* __restrict__ m2, int m2ld,
                            int br, const float* __restrict__ g, const float* __restrict__ bb,
                            float* __restrict__ r, half_t* __restrict__ rln) {
  __shared__ float sbuf[32];
  int row = blockIdx.x;  // b*196 + l
  int b = row / 196, l = row % 196;
  int c = threadIdx.x;   // 0..127
  float v = x[(size_t)row * 128 + c] + m2[((size_t)(b * 128 + c)) * m2ld + br * 196 + l];
  r[(size_t)row * 128 + c] = v;
  float mu = blk_sum(v, sbuf) * (1.f / 128.f);
  float d = v - mu;
  float inv = rsqrtf(blk_sum(d * d, sbuf) * (1.f / 128.f) + 1e-6f);
  rln[(size_t)row * 128 + c] = (half_t)(d * inv * g[c] + bb[c]);
}

// ---------------- fold conv_b + BN into scale/shift ----------------
__global__ void bn_pre(const float* __restrict__ g, const float* __restrict__ b,
                       const float* __restrict__ rm, const float* __restrict__ rv,
                       const float* __restrict__ cb, float* __restrict__ sc,
                       float* __restrict__ sh, int O) {
  int o = blockIdx.x * blockDim.x + threadIdx.x;
  if (o >= O) return;
  float s = g[o] * rsqrtf(rv[o] + 1e-5f);
  sc[o] = s;
  sh[o] = (cb[o] - rm[o]) * s + b[o];
}

// ---------------- nearest-neighbor upsample write (bandwidth floor) ----------------
__global__ void upsample(const float* __restrict__ lo, float* __restrict__ out,
                         int O, int s, int total) {
  int t = blockIdx.x * blockDim.x + threadIdx.x;
  if (t >= total) return;
  int HW2 = 14 * s;
  int w = t % HW2;
  int h = (t / HW2) % HW2;
  int o = (t / (HW2 * HW2)) % O;
  int b = t / (HW2 * HW2 * O);
  out[t] = lo[((size_t)(b * 196 + (h / s) * 14 + (w / s))) * O + o];
}

// ---------------- host helpers ----------------
static inline void gemm(hipStream_t st, const half_t* A, const half_t* B, float* C, half_t* Ch,
                        int M, int Nld, int K, int Nv, const float* bias, const float* res,
                        int act, const float* bnsc, const float* bnsh) {
  dim3 g(Nld / 64, M / 32), b(64);
  hipLaunchKernelGGL(gemm_wmma, g, b, 0, st, A, B, C, Ch, M, Nld, K, Nv, bias, res, act, bnsc, bnsh);
}
static inline void cvt(hipStream_t st, const float* src, int sld, int sr, int sc,
                       half_t* dst, int dld, int dr) {
  int tot = dld * dr;
  hipLaunchKernelGGL(convert_pad, dim3((tot + 255) / 256), dim3(256), 0, st,
                     src, sld, sr, sc, dst, dld, dr);
}
static inline void zfill(hipStream_t st, half_t* dst, int ld, int c0, int rows) {
  int tot = (ld - c0) * rows;
  hipLaunchKernelGGL(zfill_h, dim3((tot + 255) / 256), dim3(256), 0, st, dst, ld, c0, rows);
}

extern "C" void kernel_launch(void* const* d_in, const int* in_sizes, int n_in,
                              void* d_out, int out_size, void* d_ws, size_t ws_size,
                              hipStream_t stream) {
  (void)in_sizes; (void)n_in; (void)out_size; (void)ws_size;
  const float* x[4];
  for (int i = 0; i < 4; ++i) x[i] = (const float*)d_in[i];
  // params flattened as a JAX pytree (dict keys sorted alphabetically)
  const float* cm_Alog = (const float*)d_in[4];
  const float* cm_D    = (const float*)d_in[5];
  const float* cm_cb   = (const float*)d_in[6];
  const float* cm_cw   = (const float*)d_in[7];
  const float* cm_dtb  = (const float*)d_in[8];
  const float* cm_dtw  = (const float*)d_in[9];
  const float* cm_inw  = (const float*)d_in[10];
  const float* cm_outw = (const float*)d_in[11];
  const float* cm_xw   = (const float*)d_in[12];
  const float* cnb     = (const float*)d_in[13];
  const float* cng     = (const float*)d_in[14];
  const float *fc1b[4], *fc1w[4], *fc2b[4], *fc2w[4];
  for (int i = 0; i < 4; ++i) {
    fc1b[i] = (const float*)d_in[15 + 4 * i];
    fc1w[i] = (const float*)d_in[16 + 4 * i];
    fc2b[i] = (const float*)d_in[17 + 4 * i];
    fc2w[i] = (const float*)d_in[18 + 4 * i];
  }
  const float* ffnb = (const float*)d_in[31];
  const float* ffng = (const float*)d_in[32];
  const float *bnb[4], *bng[4], *bnrm[4], *bnrv[4], *rcb[4], *rcw[4];
  for (int i = 0; i < 4; ++i) {
    int base = 33 + 6 * i;
    bnb[i]  = (const float*)d_in[base + 0];
    bng[i]  = (const float*)d_in[base + 1];
    bnrm[i] = (const float*)d_in[base + 2];
    bnrv[i] = (const float*)d_in[base + 3];
    rcb[i]  = (const float*)d_in[base + 4];
    rcw[i]  = (const float*)d_in[base + 5];
  }
  const float* sm_Alog = (const float*)d_in[57];
  const float* sm_D    = (const float*)d_in[58];
  const float* sm_cb   = (const float*)d_in[59];
  const float* sm_cw   = (const float*)d_in[60];
  const float* sm_dtb  = (const float*)d_in[61];
  const float* sm_dtw  = (const float*)d_in[62];
  const float* sm_inw  = (const float*)d_in[63];
  const float* sm_outw = (const float*)d_in[64];
  const float* sm_xw   = (const float*)d_in[65];
  const float* snb     = (const float*)d_in[66];
  const float* sng     = (const float*)d_in[67];

  // bump allocator over d_ws
  char* wsb = (char*)d_ws;
  size_t off = 0;
  auto alloc = [&](size_t bytes) -> void* {
    void* p = wsb + off;
    off = (off + bytes + 255) & ~(size_t)255;
    return p;
  };
  // fp16 weights (B matrices, rows padded to Nld, cols padded to K64)
  half_t* w_in1  = (half_t*)alloc((size_t)2048 * 512 * 2);
  half_t* w_xp1  = (half_t*)alloc((size_t)NXP1 * 1024 * 2);
  half_t* w_dt1  = (half_t*)alloc((size_t)1024 * DTR1P * 2);
  half_t* w_out1 = (half_t*)alloc((size_t)512 * 1024 * 2);
  half_t* w_in2  = (half_t*)alloc((size_t)3136 * KP2 * 2);
  half_t* w_xp2  = (half_t*)alloc((size_t)XLD2 * DI2P * 2);
  half_t* w_dt2  = (half_t*)alloc((size_t)DI2P * DTR2P * 2);
  half_t* w_out2 = (half_t*)alloc((size_t)KP2 * DI2P * 2);
  half_t *wf1[4], *wf2[4], *wrc[4];
  const int OC[4] = {64, 64, 128, 256};
  const int SC[4] = {16, 8, 4, 2};
  for (int i = 0; i < 4; ++i) {
    wf1[i] = (half_t*)alloc((size_t)256 * 128 * 2);
    wf2[i] = (half_t*)alloc((size_t)128 * 256 * 2);
    wrc[i] = (half_t*)alloc((size_t)OC[i] * 128 * 2);
  }
  // activations
  half_t* c1ln  = (half_t*)alloc((size_t)R1 * 512 * 2);
  float*  xz1   = (float*)alloc((size_t)R1 * 2048 * 4);
  float*  xc1   = (float*)alloc((size_t)R1 * DI1 * 4);
  half_t* xc1h  = (half_t*)alloc((size_t)R1 * DI1 * 2);
  float*  xdbl1 = (float*)alloc((size_t)R1 * NXP1 * 4);
  half_t* dtr1h = (half_t*)alloc((size_t)R1 * DTR1P * 2);
  float*  dt1   = (float*)alloc((size_t)R1 * DI1 * 4);
  half_t* y1h   = (half_t*)alloc((size_t)R1 * DI1 * 2);
  float*  m1    = (float*)alloc((size_t)R1 * 512 * 4);
  half_t* c2ln  = (half_t*)alloc((size_t)R2 * KP2 * 2);
  float*  xz2   = (float*)alloc((size_t)R2 * 3136 * 4);
  float*  xc2   = (float*)alloc((size_t)R2 * DI2 * 4);
  half_t* xc2h  = (half_t*)alloc((size_t)R2 * DI2P * 2);
  float*  xdbl2 = (float*)alloc((size_t)R2 * XLD2 * 4);
  half_t* dtr2h = (half_t*)alloc((size_t)R2 * DTR2P * 2);
  float*  dt2   = (float*)alloc((size_t)R2 * DI2P * 4);
  half_t* y2h   = (half_t*)alloc((size_t)R2 * DI2P * 2);
  float*  m2    = (float*)alloc((size_t)R2 * KP2 * 4);
  float*  rbuf  = (float*)alloc((size_t)R1 * 128 * 4);
  half_t* rln   = (half_t*)alloc((size_t)R1 * 128 * 2);
  float*  h1    = (float*)alloc((size_t)R1 * 256 * 4);
  half_t* h1h   = (half_t*)alloc((size_t)R1 * 256 * 2);
  float*  ybuf  = (float*)alloc((size_t)R1 * 128 * 4);
  half_t* ybh   = (half_t*)alloc((size_t)R1 * 128 * 2);
  float*  lo    = (float*)alloc((size_t)R1 * 256 * 4);
  float*  bnsc  = (float*)alloc(256 * 4);
  float*  bnsh  = (float*)alloc(256 * 4);

  // --- weight conversions (fp32 -> fp16, zero padded) ---
  cvt(stream, cm_inw,  512,  2048, 512,  w_in1,  512,   2048);
  cvt(stream, cm_xw,   1024, NXP1, 1024, w_xp1,  1024,  NXP1);
  cvt(stream, cm_dtw,  DTR1, 1024, DTR1, w_dt1,  DTR1P, 1024);
  cvt(stream, cm_outw, 1024, 512,  1024, w_out1, 1024,  512);
  cvt(stream, sm_inw,  DM2,  3136, DM2,  w_in2,  KP2,   3136);
  cvt(stream, sm_xw,   DI2,  81,   DI2,  w_xp2,  DI2P,  XLD2);
  cvt(stream, sm_dtw,  DTR2, DI2,  DTR2, w_dt2,  DTR2P, DI2P);
  cvt(stream, sm_outw, DI2,  DM2,  DI2,  w_out2, DI2P,  KP2);
  for (int i = 0; i < 4; ++i) {
    cvt(stream, fc1w[i], 128, 256, 128, wf1[i], 128, 256);
    cvt(stream, fc2w[i], 256, 128, 256, wf2[i], 256, 128);
    cvt(stream, rcw[i],  128, OC[i], 128, wrc[i], 128, OC[i]);
  }

  // --- stage 1: channel mamba (d_model=512, di=1024, L=196) ---
  hipLaunchKernelGGL(ln_concat, dim3(R1), dim3(256), 0, stream,
                     x[0], x[1], x[2], x[3], cng, cnb, c1ln);
  gemm(stream, c1ln, w_in1, xz1, nullptr, R1, 2048, 512, 2048, nullptr, nullptr, 0, nullptr, nullptr);
  {
    int tot = BB * L1 * DI1;
    hipLaunchKernelGGL(conv_silu, dim3((tot + 255) / 256), dim3(256), 0, stream,
                       xz1, 2048, DI1, L1, cm_cw, cm_cb, xc1, xc1h, DI1, tot);
  }
  gemm(stream, xc1h, w_xp1, xdbl1, nullptr, R1, NXP1, DI1, NXP1, nullptr, nullptr, 0, nullptr, nullptr);
  cvt(stream, xdbl1, NXP1, R1, DTR1, dtr1h, DTR1P, R1);
  gemm(stream, dtr1h, w_dt1, dt1, nullptr, R1, DI1, DTR1P, DI1, cm_dtb, nullptr, 1, nullptr, nullptr);
  {
    int tot = BB * DI1;
    hipLaunchKernelGGL(sel_scan, dim3((tot + 255) / 256), dim3(256), 0, stream,
                       xc1, dt1, DI1, xdbl1, NXP1, DTR1, DTR1 + NST,
                       cm_Alog, cm_D, xz1, 2048, DI1, y1h, DI1, DI1, L1, BB);
  }
  gemm(stream, y1h, w_out1, m1, nullptr, R1, 512, DI1, 512, nullptr, nullptr, 0, nullptr, nullptr);

  // --- stage 2: spatial mamba (d_model=784, di=1568, L=128) ---
  hipLaunchKernelGGL(ln_c2, dim3(R2), dim3(256), 0, stream, m1, sng, snb, c2ln);
  gemm(stream, c2ln, w_in2, xz2, nullptr, R2, 3136, KP2, 3136, nullptr, nullptr, 0, nullptr, nullptr);
  {
    int tot = BB * L2 * DI2;
    hipLaunchKernelGGL(conv_silu, dim3((tot + 255) / 256), dim3(256), 0, stream,
                       xz2, 3136, DI2, L2, sm_cw, sm_cb, xc2, xc2h, DI2P, tot);
  }
  zfill(stream, xc2h, DI2P, DI2, R2);
  gemm(stream, xc2h, w_xp2, xdbl2, nullptr, R2, XLD2, DI2P, XLD2, nullptr, nullptr, 0, nullptr, nullptr);
  cvt(stream, xdbl2, XLD2, R2, DTR2, dtr2h, DTR2P, R2);
  gemm(stream, dtr2h, w_dt2, dt2, nullptr, R2, DI2P, DTR2P, DI2, sm_dtb, nullptr, 1, nullptr, nullptr);
  {
    int tot = BB * DI2;
    hipLaunchKernelGGL(sel_scan, dim3((tot + 255) / 256), dim3(256), 0, stream,
                       xc2, dt2, DI2P, xdbl2, XLD2, DTR2, DTR2 + NST,
                       sm_Alog, sm_D, xz2, 3136, DI2, y2h, DI2P, DI2, L2, BB);
  }
  zfill(stream, y2h, DI2P, DI2, R2);
  gemm(stream, y2h, w_out2, m2, nullptr, R2, KP2, DI2P, KP2, nullptr, nullptr, 0, nullptr, nullptr);

  // --- stage 3: per-branch MLP + low-res conv/BN/ReLU + upsample ---
  size_t out_off = 0;
  for (int i = 0; i < 4; ++i) {
    hipLaunchKernelGGL(branch_prep, dim3(R1), dim3(128), 0, stream,
                       x[i], m2, KP2, i, ffng + 128 * i, ffnb + 128 * i, rbuf, rln);
    gemm(stream, rln, wf1[i], h1, h1h, R1, 256, 128, 256, fc1b[i], nullptr, 2, nullptr, nullptr);
    gemm(stream, h1h, wf2[i], ybuf, ybh, R1, 128, 256, 128, fc2b[i], rbuf, 0, nullptr, nullptr);
    hipLaunchKernelGGL(bn_pre, dim3(1), dim3(256), 0, stream,
                       bng[i], bnb[i], bnrm[i], bnrv[i], rcb[i], bnsc, bnsh, OC[i]);
    gemm(stream, ybh, wrc[i], lo, nullptr, R1, OC[i], 128, OC[i], nullptr, nullptr, 3, bnsc, bnsh);
    int HW2 = 14 * SC[i];
    int tot = BB * OC[i] * HW2 * HW2;
    hipLaunchKernelGGL(upsample, dim3((tot + 255) / 256), dim3(256), 0, stream,
                       lo, (float*)d_out + out_off, OC[i], SC[i], tot);
    out_off += (size_t)tot;
  }
}
